// Transducer_5317169512732
// MI455X (gfx1250) — matrix operand order repaired and checked by reading, not compiled, not run on hardware
//
#include <hip/hip_runtime.h>
#include <hip/hip_bf16.h>

typedef __attribute__((ext_vector_type(2)))  float    v2f;
typedef __attribute__((ext_vector_type(8)))  float    v8f;
typedef __attribute__((ext_vector_type(8)))  _Float16 v8h;
typedef __attribute__((ext_vector_type(16))) _Float16 v16h;

#define INNER  512
#define VOCAB  2048
#define T_     256
#define U_     128
#define HS     520   // padded LDS row stride in halves (520%64==8 halves = 4 dwords -> conflict-free-ish b128)

// ---------------------------------------------------------------------------
// Phase 1a/1b: eproj = enc @ W1[:512]        (M=1024)
//              dproj = dec @ W1[512:] + b1   (M=512)
// Full-precision f32 WMMA (16x16x4) since these feed tanh.
// One wave per 16x16 tile, 8 waves/block -> block covers 16 rows x 128 cols.
// ---------------------------------------------------------------------------
__global__ __launch_bounds__(256)
void proj_gemm_f32(const float* __restrict__ A, const float* __restrict__ W,
                   const float* __restrict__ bias, float* __restrict__ C) {
    const int wave = threadIdx.x >> 5;
    const int lane = threadIdx.x & 31;
    const int m    = lane & 15;
    const int sel  = lane >> 4;          // 0: K0..1 / rows 0..7, 1: K2..3 / rows 8..15
    const int kOff = sel * 2;
    const int r0   = blockIdx.x * 16;
    const int n0   = (blockIdx.y * 8 + wave) * 16;

    const float* arow = A + (size_t)(r0 + m) * INNER + kOff;   // A[r0+m][kOff + k0 ..]
    const float* wcol = W + n0 + m;                            // W[k][n0+m]

    v8f acc = {0.f, 0.f, 0.f, 0.f, 0.f, 0.f, 0.f, 0.f};
#pragma unroll 4
    for (int k0 = 0; k0 < INNER; k0 += 4) {
        v2f a = *(const v2f*)(arow + k0);                      // A[m][k0+kOff], A[m][k0+kOff+1]
        v2f b;
        b.x = wcol[(size_t)(k0 + kOff)     * INNER];           // B[k0+kOff  ][n]
        b.y = wcol[(size_t)(k0 + kOff + 1) * INNER];           // B[k0+kOff+1][n]
        acc = __builtin_amdgcn_wmma_f32_16x16x4_f32(false, a, false, b,
                                                    (short)0, acc, false, false);
    }
    const float bv = bias ? bias[n0 + m] : 0.0f;
#pragma unroll
    for (int r = 0; r < 8; ++r)
        C[(size_t)(r0 + sel * 8 + r) * INNER + n0 + m] = acc[r] + bv;
}

// ---------------------------------------------------------------------------
// Phase 1c: W2T[n][k] = (f16) W2[k][n]   (2048 x 512 f16, 2MB, L2-resident)
// ---------------------------------------------------------------------------
__global__ __launch_bounds__(256)
void transpose_w2(const float* __restrict__ W2, _Float16* __restrict__ W2T) {
    __shared__ float tile[32][33];
    const int n0 = blockIdx.x * 32;
    const int k0 = blockIdx.y * 32;
    const int tx = threadIdx.x;      // 0..31
    const int ty = threadIdx.y;      // 0..7
#pragma unroll
    for (int i = 0; i < 32; i += 8)
        tile[ty + i][tx] = W2[(size_t)(k0 + ty + i) * VOCAB + n0 + tx];
    __syncthreads();
#pragma unroll
    for (int i = 0; i < 32; i += 8)
        W2T[(size_t)(n0 + ty + i) * INNER + k0 + tx] = (_Float16)tile[tx][ty + i];
}

// ---------------------------------------------------------------------------
// Phase 2: one block per (b,t). Build h = tanh(e + d) as f16 in LDS
// (128 x 512, padded), then h @ W2^T with v_wmma_f32_16x16x32_f16.
// 512 threads = 16 waves; wave blocking = 2 row-tiles x 4 col-tiles.
// ---------------------------------------------------------------------------
__global__ __launch_bounds__(512)
void joint_wmma_f16(const float* __restrict__ eproj, const float* __restrict__ dproj,
                    const _Float16* __restrict__ w2t, const float* __restrict__ b2,
                    float* __restrict__ out) {
    __shared__ _Float16 hsh[U_ * HS];                 // 128 * 520 * 2 = 130 KB
    const int bt  = blockIdx.x;                        // b*T + t
    const int tid = threadIdx.x;

    // ---- build h tile: thread 'tid' owns column 'tid' for all 128 u-rows ----
    const float  e  = eproj[(size_t)bt * INNER + tid];
    const float* dp = dproj + (size_t)(bt >> 8) * U_ * INNER + tid;   // b = bt / T_
#pragma unroll 4
    for (int r = 0; r < U_; ++r)
        hsh[r * HS + tid] = (_Float16)tanhf(e + dp[(size_t)r * INNER]);
    __syncthreads();

    const int wave = tid >> 5;
    const int lane = tid & 31;
    const int m    = lane & 15;
    const int sel  = lane >> 4;
    float* outbt   = out + (size_t)bt * U_ * VOCAB;

    // 4 row-groups (32 rows) x 32 col-groups (64 cols) = 128 pairs / 16 waves = 8 each
    for (int q = 0; q < 8; ++q) {
        const int p   = wave * 8 + q;
        const int rtg = p & 3;         // row group: rows [rtg*32, rtg*32+32)
        const int cg  = p >> 2;        // col group: cols [cg*64, cg*64+64)

        const _Float16* hrow0 = hsh + (rtg * 32 + m) * HS + sel * 16;
        const _Float16* hrow1 = hrow0 + 16 * HS;
        const _Float16* bbase = w2t + (size_t)(cg * 64 + m) * INNER + sel * 16;

        v8f acc[2][4];
#pragma unroll
        for (int rt = 0; rt < 2; ++rt)
#pragma unroll
            for (int j = 0; j < 4; ++j)
                acc[rt][j] = v8f{0.f, 0.f, 0.f, 0.f, 0.f, 0.f, 0.f, 0.f};

#pragma unroll 2
        for (int k0 = 0; k0 < INNER; k0 += 32) {
            // A tiles (16x32 f16 each): lane m holds K [k0+sel*16, +16) contiguous
            v8h a0lo = *(const v8h*)(hrow0 + k0);
            v8h a0hi = *(const v8h*)(hrow0 + k0 + 8);
            v8h a1lo = *(const v8h*)(hrow1 + k0);
            v8h a1hi = *(const v8h*)(hrow1 + k0 + 8);
            v16h a0, a1;
#pragma unroll
            for (int i = 0; i < 8; ++i) {
                a0[i] = a0lo[i]; a0[i + 8] = a0hi[i];
                a1[i] = a1lo[i]; a1[i + 8] = a1hi[i];
            }
            // B tiles (32x16 f16): lane holds 16 consecutive K of one column (32B contiguous)
            v16h bb[4];
#pragma unroll
            for (int j = 0; j < 4; ++j)
                bb[j] = *(const v16h*)(bbase + (size_t)j * 16 * INNER + k0);
#pragma unroll
            for (int j = 0; j < 4; ++j) {
                acc[0][j] = __builtin_amdgcn_wmma_f32_16x16x32_f16(
                    false, a0, false, bb[j], (short)0, acc[0][j], false, false);
                acc[1][j] = __builtin_amdgcn_wmma_f32_16x16x32_f16(
                    false, a1, false, bb[j], (short)0, acc[1][j], false, false);
            }
        }

        // epilogue: + b2, streaming (non-temporal) stores
#pragma unroll
        for (int j = 0; j < 4; ++j) {
            const int   col = cg * 64 + j * 16 + m;
            const float b2v = b2[col];
#pragma unroll
            for (int rt = 0; rt < 2; ++rt) {
                const int rowbase = rtg * 32 + rt * 16 + sel * 8;
#pragma unroll
                for (int r = 0; r < 8; ++r)
                    __builtin_nontemporal_store(acc[rt][j][r] + b2v,
                        outbt + (size_t)(rowbase + r) * VOCAB + col);
            }
        }
    }
}

// ---------------------------------------------------------------------------
extern "C" void kernel_launch(void* const* d_in, const int* in_sizes, int n_in,
                              void* d_out, int out_size, void* d_ws, size_t ws_size,
                              hipStream_t stream) {
    const float* enc = (const float*)d_in[0];   // (4,256,512)
    const float* dec = (const float*)d_in[1];   // (4,128,512)
    const float* W1  = (const float*)d_in[2];   // (1024,512)
    const float* b1  = (const float*)d_in[3];   // (512,)
    const float* W2  = (const float*)d_in[4];   // (512,2048)
    const float* b2  = (const float*)d_in[5];   // (2048,)
    float* out = (float*)d_out;                 // (4,256,128,2048)

    float*     eproj = (float*)d_ws;                       // 1024*512 f32 = 2MB
    float*     dproj = eproj + 1024 * INNER;               //  512*512 f32 = 1MB
    _Float16*  w2t   = (_Float16*)(dproj + 512 * INNER);   // 2048*512 f16 = 2MB

    // eproj = enc @ W1[:512]
    proj_gemm_f32<<<dim3(64, 4), 256, 0, stream>>>(enc, W1, nullptr, eproj);
    // dproj = dec @ W1[512:] + b1
    proj_gemm_f32<<<dim3(32, 4), 256, 0, stream>>>(dec, W1 + 512 * INNER, b1, dproj);
    // W2^T in f16
    transpose_w2<<<dim3(64, 16), dim3(32, 8), 0, stream>>>(W2, w2t);
    // logits
    joint_wmma_f16<<<dim3(1024), 512, 0, stream>>>(eproj, dproj, w2t, b2, out);
}
